// MultiHeadAttention_30313879176070
// MI455X (gfx1250) — matrix operand order repaired and checked by reading, not compiled, not run on hardware
//
#include <hip/hip_runtime.h>
#include <stdint.h>

#define T_SEQ   2048
#define BATCH   2
#define DMODEL  1024
#define NHEADS  16
#define DHEAD   64
#define MROWS   (BATCH * T_SEQ)   // 4096
#define RELN    (2 * T_SEQ - 1)   // 4095
#define LSTR    72                // LDS row stride (bf16 elems) for 64-wide K chunks

typedef __attribute__((ext_vector_type(16))) __bf16 v16bf;
typedef __attribute__((ext_vector_type(8)))  float  v8f;

union FragBF { v16bf v; uint4 q[2]; };

__device__ __forceinline__ unsigned short f2bf(float f) {
  union { float f; unsigned u; } x; x.f = f;
  unsigned r = x.u + 0x7fffu + ((x.u >> 16) & 1u);   // round-to-nearest-even
  return (unsigned short)(r >> 16);
}

__device__ __forceinline__ v8f wmma_bf16(const v16bf& a, const v16bf& b, v8f c) {
  return __builtin_amdgcn_wmma_f32_16x16x32_bf16(false, a, false, b, (short)0, c, false, false);
}

// async global -> LDS 16-byte copy (per active lane), tracked by ASYNCcnt
__device__ __forceinline__ void async_b128(unsigned lds_off, const void* gptr) {
  asm volatile("global_load_async_to_lds_b128 %0, %1, off"
               :: "v"(lds_off), "v"((unsigned long long)gptr) : "memory");
}
__device__ __forceinline__ void wait_async0()  { asm volatile("s_wait_asynccnt 0x0" ::: "memory"); }
__device__ __forceinline__ void wait_async12() { asm volatile("s_wait_asynccnt 0xc" ::: "memory"); }

// ---------------- fp32 -> bf16 convert ----------------
__global__ void cvt_kernel(const float* __restrict__ src,
                           unsigned short* __restrict__ dst, int n) {
  int i = blockIdx.x * blockDim.x + threadIdx.x;
  if (i < n) dst[i] = f2bf(src[i]);
}

// ---------------- T5 relative bias table: brel[h][rel + T-1] ----------------
__global__ void bias_kernel(const float* __restrict__ table,   // [32,16]
                            float* __restrict__ brel) {        // [16,4095]
  int i = blockIdx.x * blockDim.x + threadIdx.x;
  if (i >= RELN * NHEADS) return;
  int ridx = i >> 4;
  int h    = i & 15;
  int rel  = ridx - (T_SEQ - 1);          // k_pos - q_pos
  int sgn  = (rel > 0) ? 1 : 0;
  int n    = (rel < 0) ? -rel : rel;
  int bucket;
  if (n < 8) {
    bucket = n;
  } else {
    float t = logf((float)n / 8.0f + 1e-6f) * (1.0f / logf(16.0f)) * 8.0f;
    int vl = 8 + (int)t;
    bucket = (vl < 15) ? vl : 15;
  }
  bucket += sgn * 16;
  brel[(size_t)h * RELN + ridx] = table[bucket * NHEADS + h];
}

// ---------------- bf16 WMMA GEMM: out[m,n] = sum_k A[m,k] * W[n,k] ----------------
// 128(M) x 64(N) per workgroup; K in 64-chunks; double-buffered async-to-LDS staging.
// MODE 0: Q  -> [B,H,T,64] bf16, scaled by 1/sqrt(64)
// MODE 1: K  -> [B,H,T,64] bf16
// MODE 2: V  -> [B,H,64,T] bf16 (transposed)
// MODE 3: fp32 -> d_out [M, 1024]
template <int MODE>
__global__ __launch_bounds__(128) void gemm_kernel(
    const unsigned short* __restrict__ A,   // [M,1024] bf16
    const unsigned short* __restrict__ W,   // [1024,1024] bf16 (row n, col k)
    void* __restrict__ Out) {
  __shared__ __align__(16) unsigned short ldsA[2][128 * LSTR];  // 2 x 18 KB
  __shared__ __align__(16) unsigned short ldsW[2][64 * LSTR];   // 2 x  9 KB
  const int tid  = threadIdx.x;
  const int lane = tid & 31;
  const int w    = tid >> 5;        // wave 0..3 -> M sub-tile
  const int col  = lane & 15;
  const int kh   = lane >> 4;
  const int n0   = blockIdx.x * 64;
  const int m0   = blockIdx.y * 128;

  v8f c[2][4];
#pragma unroll
  for (int hlf = 0; hlf < 2; ++hlf)
#pragma unroll
    for (int j = 0; j < 4; ++j) c[hlf][j] = (v8f){0.f,0.f,0.f,0.f,0.f,0.f,0.f,0.f};

  // stage one 64-wide K chunk into buffer `buf` (12 async b128 per thread)
  auto stage = [&](int k0, int buf) {
    const unsigned a0 = (unsigned)(size_t)&ldsA[buf][0];
    const unsigned w0 = (unsigned)(size_t)&ldsW[buf][0];
#pragma unroll
    for (int i = 0; i < 8; ++i) {               // A tile: 128x64
      int j = tid + i * 128;                    // 0..1023
      int row = j >> 3, seg = j & 7;
      async_b128(a0 + (unsigned)(row * LSTR + seg * 8) * 2,
                 &A[(size_t)(m0 + row) * DMODEL + k0 + seg * 8]);
    }
#pragma unroll
    for (int i = 0; i < 4; ++i) {               // W tile: 64x64
      int j = tid + i * 128;                    // 0..511
      int row = j >> 3, seg = j & 7;
      async_b128(w0 + (unsigned)(row * LSTR + seg * 8) * 2,
                 &W[(size_t)(n0 + row) * DMODEL + k0 + seg * 8]);
    }
  };

  auto compute = [&](int buf) {
#pragma unroll
    for (int cc = 0; cc < 2; ++cc) {            // two K=32 sub-chunks
      FragBF a0, a1;
      const unsigned short* ar0 = &ldsA[buf][(16 * w + col) * LSTR + 32 * cc];
      const unsigned short* ar1 = &ldsA[buf][(64 + 16 * w + col) * LSTR + 32 * cc];
      a0.q[0] = *(const uint4*)(ar0 + 8 * kh);
      a0.q[1] = *(const uint4*)(ar0 + 16 + 8 * kh);
      a1.q[0] = *(const uint4*)(ar1 + 8 * kh);
      a1.q[1] = *(const uint4*)(ar1 + 16 + 8 * kh);
#pragma unroll
      for (int j = 0; j < 4; ++j) {
        const unsigned short* brow = &ldsW[buf][(16 * j + col) * LSTR + 32 * cc + 8 * kh];
        FragBF bf;
        bf.q[0] = *(const uint4*)(brow);
        bf.q[1] = *(const uint4*)(brow + 16);
        c[0][j] = wmma_bf16(a0.v, bf.v, c[0][j]);
        c[1][j] = wmma_bf16(a1.v, bf.v, c[1][j]);
      }
    }
  };

  // double-buffered pipeline over 16 K-chunks
  stage(0, 0);
#pragma unroll 1
  for (int i = 0; i + 1 < 16; ++i) {
    stage((i + 1) * 64, (i + 1) & 1);   // issue next tile's DMA
    wait_async12();                     // previous 12 (current tile) have landed
    __syncthreads();
    compute(i & 1);
    __syncthreads();                    // buffer reusable after this
  }
  wait_async0();
  __syncthreads();
  compute(1);                           // chunk 15 lives in buffer 1

#pragma unroll
  for (int hlf = 0; hlf < 2; ++hlf) {
#pragma unroll
    for (int j = 0; j < 4; ++j) {
#pragma unroll
      for (int g = 0; g < 8; ++g) {
        int m = m0 + 64 * hlf + 16 * w + g + 8 * kh;
        int n = n0 + 16 * j + col;
        float val = c[hlf][j][g];
        if (MODE == 3) {
          ((float*)Out)[(size_t)m * DMODEL + n] = val;
        } else {
          int b = m >> 11, t = m & (T_SEQ - 1);
          int h = n >> 6,  d = n & 63;
          unsigned short bv = f2bf(MODE == 0 ? val * 0.125f : val);
          unsigned short* O = (unsigned short*)Out;
          if (MODE == 2) O[((size_t)(b * NHEADS + h) * DHEAD + d) * T_SEQ + t] = bv;
          else           O[((size_t)(b * NHEADS + h) * T_SEQ + t) * DHEAD + d] = bv;
        }
      }
    }
  }
}

// ---------------- flash attention: 1 wave per 16-query tile ----------------
__global__ __launch_bounds__(32) void attn_kernel(
    const unsigned short* __restrict__ Qb,    // [B,H,T,64] bf16 (pre-scaled)
    const unsigned short* __restrict__ Kb,    // [B,H,T,64] bf16
    const unsigned short* __restrict__ Vtb,   // [B,H,64,T] bf16
    const float* __restrict__ biasRel,        // [H, 4095]
    const unsigned char* __restrict__ kpm,    // [B, T] bool
    unsigned short* __restrict__ Ctx) {       // [M, 1024] bf16
  __shared__ __align__(16) unsigned short ldsP[16 * 40];
  const int lane = threadIdx.x & 31;
  const int col  = lane & 15;
  const int kh   = lane >> 4;
  const int q0 = blockIdx.x * 16;
  const int h  = blockIdx.y;
  const int b  = blockIdx.z;

  const size_t bh = (size_t)(b * NHEADS + h);
  const unsigned short* Qp = Qb  + bh * T_SEQ * DHEAD;
  const unsigned short* Kp = Kb  + bh * T_SEQ * DHEAD;
  const unsigned short* Vp = Vtb + bh * DHEAD * T_SEQ;
  const float* brel = biasRel + (size_t)h * RELN;
  const unsigned char* kp = kpm + (size_t)b * T_SEQ;

  // Q A-fragments: two K=32 chunks over Dh=64
  FragBF aq[2];
  {
    const unsigned short* qrow = Qp + (size_t)(q0 + col) * DHEAD;
#pragma unroll
    for (int c2 = 0; c2 < 2; ++c2) {
      aq[c2].q[0] = *(const uint4*)(qrow + 32 * c2 + 16 * kh);
      aq[c2].q[1] = *(const uint4*)(qrow + 32 * c2 + 16 * kh + 8);
    }
  }

  float mrow[8], lrow[8];
  v8f acc[4];
#pragma unroll
  for (int g = 0; g < 8; ++g) { mrow[g] = -3.0e38f; lrow[g] = 0.0f; }
#pragma unroll
  for (int c2 = 0; c2 < 4; ++c2) acc[c2] = (v8f){0.f,0.f,0.f,0.f,0.f,0.f,0.f,0.f};

  const int kend = q0 + 16;                      // causal: keys < q0+16
  for (int kt = 0; kt < kend; kt += 32) {
    // prefetch next tile's K / V rows while this one computes
    __builtin_prefetch(Kp + (size_t)(kt + 32 + col) * DHEAD, 0, 3);
    __builtin_prefetch(Vp + (size_t)col * T_SEQ + kt + 32, 0, 3);
    // per-lane key coordinates (fixed over the row loop)
    const int key0 = kt + col;
    const int key1 = kt + 16 + col;
    const bool kp0 = (kp[key0] != 0);
    const bool kp1 = (kp[key1] != 0);
    // bias gathers are always in-bounds (rel in [-2047,46]) -> load
    // unconditionally, batched, so they overlap the score WMMAs.
    const int bbase0 = key0 - q0 - 8 * kh + (T_SEQ - 1);
    const int bbase1 = key1 - q0 - 8 * kh + (T_SEQ - 1);
    float bias0[8], bias1[8];
#pragma unroll
    for (int g = 0; g < 8; ++g) {
      bias0[g] = brel[bbase0 - g];
      bias1[g] = brel[bbase1 - g];
    }
    // ---- scores S = Q K^T (two 16x16 C-frags for 32 keys) ----
    v8f s[2];
#pragma unroll
    for (int n = 0; n < 2; ++n) {
      const unsigned short* krow = Kp + (size_t)(kt + 16 * n + col) * DHEAD;
      FragBF bk0, bk1;
      bk0.q[0] = *(const uint4*)(krow + 16 * kh);
      bk0.q[1] = *(const uint4*)(krow + 16 * kh + 8);
      bk1.q[0] = *(const uint4*)(krow + 32 + 16 * kh);
      bk1.q[1] = *(const uint4*)(krow + 32 + 16 * kh + 8);
      v8f z = (v8f){0.f,0.f,0.f,0.f,0.f,0.f,0.f,0.f};
      z = wmma_bf16(aq[0].v, bk0.v, z);
      z = wmma_bf16(aq[1].v, bk1.v, z);
      s[n] = z;
    }
    // ---- bias + masks + online softmax stats (rows live per VGPR g) ----
    float alpha[8];
#pragma unroll
    for (int g = 0; g < 8; ++g) {
      const int qr = q0 + g + 8 * kh;
      float v0 = s[0][g] + bias0[g];
      float v1 = s[1][g] + bias1[g];
      v0 = ((key0 <= qr) && kp0) ? v0 : -3.0e38f;   // plain cndmask, no branch
      v1 = ((key1 <= qr) && kp1) ? v1 : -3.0e38f;
      float rm = fmaxf(v0, v1);
#pragma unroll
      for (int msk = 1; msk < 16; msk <<= 1) rm = fmaxf(rm, __shfl_xor(rm, msk, 32));
      float mn = fmaxf(mrow[g], rm);
      alpha[g] = __expf(mrow[g] - mn);
      float p0 = __expf(v0 - mn);
      float p1 = __expf(v1 - mn);
      float rs = p0 + p1;
#pragma unroll
      for (int msk = 1; msk < 16; msk <<= 1) rs += __shfl_xor(rs, msk, 32);
      lrow[g] = lrow[g] * alpha[g] + rs;
      mrow[g] = mn;
      // P tile -> LDS (row-major 16x32, stride 40) for C->A re-swizzle
      ldsP[(g + 8 * kh) * 40 + col]      = f2bf(p0);
      ldsP[(g + 8 * kh) * 40 + 16 + col] = f2bf(p1);
    }
    // rescale ctx accumulators
#pragma unroll
    for (int c2 = 0; c2 < 4; ++c2)
#pragma unroll
      for (int g = 0; g < 8; ++g) acc[c2][g] *= alpha[g];
    // ---- P as A-fragment (16 rows x 32 keys) ----
    FragBF ap;
    const unsigned short* prow = &ldsP[col * 40];
    ap.q[0] = *(const uint4*)(prow + 16 * kh);
    ap.q[1] = *(const uint4*)(prow + 16 * kh + 8);
    // ---- ctx += P @ V  (Vt rows are contiguous over keys) ----
#pragma unroll
    for (int c2 = 0; c2 < 4; ++c2) {
      const unsigned short* vrow = Vp + (size_t)(16 * c2 + col) * T_SEQ + kt + 16 * kh;
      FragBF bv;
      bv.q[0] = *(const uint4*)(vrow);
      bv.q[1] = *(const uint4*)(vrow + 8);
      acc[c2] = wmma_bf16(ap.v, bv.v, acc[c2]);
    }
  }

  // ---- normalize and emit ctx in [B,T, H*64] layout (bf16) ----
  float rinv[8];
#pragma unroll
  for (int g = 0; g < 8; ++g) rinv[g] = 1.0f / lrow[g];
#pragma unroll
  for (int c2 = 0; c2 < 4; ++c2) {
#pragma unroll
    for (int g = 0; g < 8; ++g) {
      int t = q0 + g + 8 * kh;
      float val = acc[c2][g] * rinv[g];
      Ctx[(size_t)(b * T_SEQ + t) * DMODEL + h * DHEAD + 16 * c2 + col] = f2bf(val);
    }
  }
}

// ---------------- host-side orchestration ----------------
extern "C" void kernel_launch(void* const* d_in, const int* in_sizes, int n_in,
                              void* d_out, int out_size, void* d_ws, size_t ws_size,
                              hipStream_t stream) {
  (void)in_sizes; (void)n_in; (void)out_size; (void)ws_size;
  const float* x_q  = (const float*)d_in[0];
  const float* x_kv = (const float*)d_in[1];
  const float* Wq   = (const float*)d_in[2];
  const float* Wk   = (const float*)d_in[3];
  const float* Wv   = (const float*)d_in[4];
  const float* Wo   = (const float*)d_in[5];
  const float* tbl  = (const float*)d_in[6];
  // d_in[7] = attn_mask (causal tril) -- applied analytically in-kernel
  const unsigned char* kpm = (const unsigned char*)d_in[8];

  char* ws = (char*)d_ws;
  size_t off = 0;
  auto alloc = [&](size_t bytes) -> char* {
    char* p = ws + off; off += (bytes + 255) & ~(size_t)255; return p;
  };
  const size_t MD = (size_t)MROWS * DMODEL;   // 4.19M elems
  const size_t DD = (size_t)DMODEL * DMODEL;  // 1.05M elems
  unsigned short* Xqb  = (unsigned short*)alloc(MD * 2);
  unsigned short* Xkb  = (unsigned short*)alloc(MD * 2);
  unsigned short* Wqb  = (unsigned short*)alloc(DD * 2);
  unsigned short* Wkb  = (unsigned short*)alloc(DD * 2);
  unsigned short* Wvb  = (unsigned short*)alloc(DD * 2);
  unsigned short* Wob  = (unsigned short*)alloc(DD * 2);
  unsigned short* Qb   = (unsigned short*)alloc(MD * 2);
  unsigned short* Kb   = (unsigned short*)alloc(MD * 2);
  unsigned short* Vtb  = (unsigned short*)alloc(MD * 2);
  unsigned short* Ctxb = (unsigned short*)alloc(MD * 2);
  float*          brel = (float*)alloc((size_t)RELN * NHEADS * 4);

  const int thr = 256;
  cvt_kernel<<<(int)((MD + thr - 1) / thr), thr, 0, stream>>>(x_q,  Xqb, (int)MD);
  cvt_kernel<<<(int)((MD + thr - 1) / thr), thr, 0, stream>>>(x_kv, Xkb, (int)MD);
  cvt_kernel<<<(int)((DD + thr - 1) / thr), thr, 0, stream>>>(Wq, Wqb, (int)DD);
  cvt_kernel<<<(int)((DD + thr - 1) / thr), thr, 0, stream>>>(Wk, Wkb, (int)DD);
  cvt_kernel<<<(int)((DD + thr - 1) / thr), thr, 0, stream>>>(Wv, Wvb, (int)DD);
  cvt_kernel<<<(int)((DD + thr - 1) / thr), thr, 0, stream>>>(Wo, Wob, (int)DD);
  bias_kernel<<<(RELN * NHEADS + thr - 1) / thr, thr, 0, stream>>>(tbl, brel);

  dim3 gg(DMODEL / 64, MROWS / 128);
  gemm_kernel<0><<<gg, 128, 0, stream>>>(Xqb, Wqb, Qb);
  gemm_kernel<1><<<gg, 128, 0, stream>>>(Xkb, Wkb, Kb);
  gemm_kernel<2><<<gg, 128, 0, stream>>>(Xkb, Wvb, Vtb);

  dim3 ga(T_SEQ / 16, NHEADS, BATCH);
  attn_kernel<<<ga, 32, 0, stream>>>(Qb, Kb, Vtb, brel, kpm, Ctxb);

  gemm_kernel<3><<<gg, 128, 0, stream>>>(Ctxb, Wob, d_out);
}
